// TAAMSG_67619965108634
// MI455X (gfx1250) — compile-verified, matrix operand order checked by reading, not run on hardware
//
#include <hip/hip_runtime.h>
#include <hip/hip_bf16.h>
#include <math.h>

// ---------------------------------------------------------------------------
// Sizes (fixed by the reference)
// ---------------------------------------------------------------------------
#define NN      49152
#define EH      8192
#define MM      (NN + EH)          // 57344
#define DD      256
#define HEADS   8
#define CHUNK   1024
#define DH      32                 // DD / HEADS

typedef __attribute__((ext_vector_type(16))) __bf16 v16bf;
typedef __attribute__((ext_vector_type(8)))  float  v8f;

// k-offset pattern shared by the 16-bit A fragment (lane = row) and the
// column-major B fragment (lane = col): element j of the v16bf lives at
//   k = 8*(lane>>4) + (j&7) + 16*(j>>3)
__device__ __forceinline__ int koff(int lane, int j) {
    return ((lane >> 4) << 3) + (j & 7) + ((j & 8) << 1);
}

// Load a 16-element bf16 fragment from an fp32 row. `base` must already point
// at  row_start + k_tile_base + khalf(lane).  Covers k runs [0,8) and [16,24).
__device__ __forceinline__ v16bf load_frag_f32(const float* __restrict__ base, float scale) {
    float4 a0 = *(const float4*)(base + 0);
    float4 a1 = *(const float4*)(base + 4);
    float4 a2 = *(const float4*)(base + 16);
    float4 a3 = *(const float4*)(base + 20);
    v16bf r;
    r[0]  = (__bf16)(a0.x * scale); r[1]  = (__bf16)(a0.y * scale);
    r[2]  = (__bf16)(a0.z * scale); r[3]  = (__bf16)(a0.w * scale);
    r[4]  = (__bf16)(a1.x * scale); r[5]  = (__bf16)(a1.y * scale);
    r[6]  = (__bf16)(a1.z * scale); r[7]  = (__bf16)(a1.w * scale);
    r[8]  = (__bf16)(a2.x * scale); r[9]  = (__bf16)(a2.y * scale);
    r[10] = (__bf16)(a2.z * scale); r[11] = (__bf16)(a2.w * scale);
    r[12] = (__bf16)(a3.x * scale); r[13] = (__bf16)(a3.y * scale);
    r[14] = (__bf16)(a3.z * scale); r[15] = (__bf16)(a3.w * scale);
    return r;
}

#define WMMA_BF16(a, b, c) \
    __builtin_amdgcn_wmma_f32_16x16x32_bf16(false, (a), false, (b), (short)0, (c), false, false)

// ---------------------------------------------------------------------------
// Weight packing: fp32 256x256 weight -> bf16 B-fragment layout
//   Bp[((kt*16 + nt)*32 + lane)*16 + j] = B[kt*32 + koff(lane,j)][nt*16 + (lane&15)]
// where B[k][n] = W[k*256+n] (transpose=0, used for x@W) or W[n*256+k]
// (transpose=1, used for x@W.T).  65536 threads total.
// ---------------------------------------------------------------------------
__global__ void __launch_bounds__(256)
pack_weight_kernel(const float* __restrict__ W, __bf16* __restrict__ Bp, int transpose) {
    int t    = blockIdx.x * 256 + threadIdx.x;     // 0 .. 65535
    int j    = t & 15;
    int lane = (t >> 4) & 31;
    int nt   = (t >> 9) & 15;
    int kt   = t >> 13;
    int k = kt * 32 + koff(lane, j);
    int n = nt * 16 + (lane & 15);
    float v = transpose ? W[n * 256 + k] : W[k * 256 + n];
    Bp[t] = (__bf16)v;
}

// ---------------------------------------------------------------------------
// GEMM:  C[r, col_off + c] = sum_k A[r,k] * B[k,c] (+ bias[c])
// A: rows x 256 fp32 (rows % 128 == 0), Bp: packed bf16 weight, C row stride ldc.
// Block: 256 threads = 8 waves.  The block's 128-column slice of packed B
// (8 kt x 8 nt x 32 lanes x 16 bf16 = 64KB) is staged ONCE into LDS (all 8
// waves share the same columns), then each wave computes 16 rows x 128 cols:
// per K step one A fragment (fp32 load + cvt) feeds 8 v_wmma_f32_16x16x32_bf16
// whose B fragments come from ds_load_b128.  grid = (rows/128, 2).
// ---------------------------------------------------------------------------
__global__ void __launch_bounds__(256)
gemm_wmma_kernel(const float* __restrict__ A, const __bf16* __restrict__ Bp,
                 const float* __restrict__ bias, float* __restrict__ C,
                 int ldc, int col_off) {
    extern __shared__ char smemraw[];
    __bf16* ldsB = (__bf16*)smemraw;     // [8 kt][8 nt][32 lane][16 j]

    int tid  = threadIdx.x;
    int lane = tid & 31;
    int wave = tid >> 5;
    int row0 = blockIdx.x * 128 + wave * 16;
    int nt0  = blockIdx.y * 8;           // first of 8 n-tiles (128 cols)

    // ---- stage this block's B slice into LDS (16-byte units, coalesced) ----
    for (int u = tid; u < 4096; u += 256) {
        int kt  = u >> 9;                // 512 units of 8 bf16 per kt
        int rem = u & 511;
        ((float4*)ldsB)[u] =
            *((const float4*)(Bp + (size_t)(kt * 16 + nt0) * 512) + rem);
    }
    __syncthreads();

    int khalf = (lane >> 4) << 3;
    const float* arow = A + (size_t)(row0 + (lane & 15)) * 256;
    __builtin_prefetch(arow, 0, 3);            // global_prefetch_b8
    __builtin_prefetch(arow + 128, 0, 3);

    v8f c[8];
#pragma unroll
    for (int t = 0; t < 8; ++t) c[t] = (v8f){};

    for (int kt = 0; kt < 8; ++kt) {
        v16bf af = load_frag_f32(arow + kt * 32 + khalf, 1.0f);
        const v16bf* bp = (const v16bf*)ldsB + kt * 256 + lane;
#pragma unroll
        for (int t = 0; t < 8; ++t)
            c[t] = WMMA_BF16(af, bp[t * 32], c[t]);
    }

    int nc    = lane & 15;
    int mbase = row0 + ((lane >> 4) << 3);
#pragma unroll
    for (int t = 0; t < 8; ++t) {
        float bb = bias ? bias[(nt0 + t) * 16 + nc] : 0.0f;
#pragma unroll
        for (int r = 0; r < 8; ++r) {
            C[(size_t)(mbase + r) * ldc + col_off + (nt0 + t) * 16 + nc] = c[t][r] + bb;
        }
    }
}

// ---------------------------------------------------------------------------
// SPMM scatter:  Y[row[e], :] += val[e] * X[col[e], :]   (Y pre-initialized)
// 64 lanes per edge, float4 gather + 4x global_atomic_add_f32 scatter.
// ---------------------------------------------------------------------------
__global__ void __launch_bounds__(256)
spmm_kernel(const int* __restrict__ row, const int* __restrict__ col,
            const float* __restrict__ val, const float* __restrict__ X,
            float* __restrict__ Y, int ne) {
    int t = blockIdx.x * 256 + threadIdx.x;
    int e = t >> 6;
    if (e >= ne) return;
    int f = (t & 63) << 2;
    float v = val[e];
    const float* xs = X + (size_t)col[e] * 256 + f;
    float*       yd = Y + (size_t)row[e] * 256 + f;
    float4 x = *(const float4*)xs;
    atomicAdd(yd + 0, v * x.x);
    atomicAdd(yd + 1, v * x.y);
    atomicAdd(yd + 2, v * x.z);
    atomicAdd(yd + 3, v * x.w);
}

// Y[i*256+j] = bias[j]  (or 0 if bias==null); total = rows*256 elements
__global__ void __launch_bounds__(256)
init_bias_kernel(float* __restrict__ Y, const float* __restrict__ bias, int total) {
    int t = blockIdx.x * 256 + threadIdx.x;
    if (t >= total) return;
    Y[t] = bias ? bias[t & 255] : 0.0f;
}

__global__ void __launch_bounds__(256)
relu_kernel(float* __restrict__ Y, int total) {
    int t = blockIdx.x * 256 + threadIdx.x;
    if (t >= total) return;
    float v = Y[t];
    Y[t] = v > 0.0f ? v : 0.0f;
}

// ---------------------------------------------------------------------------
// Flash attention for chunked MHA.
// QH/KH/VH: (NN,256) fp32, column = head*32 + dh.  Grid: (NN/128, HEADS),
// block 256 = 8 waves; wave owns 16 query rows of one (chunk, head).
// V is staged once per block into LDS in B-fragment layout (bf16); K
// fragments load straight from global (dh contiguous per key row).
// Softmax rows live in the WMMA C layout: row m spans 16 lanes of one half,
// reduced with 4x __shfl_xor.  P is transposed C->A layout via per-wave LDS.
// Dynamic LDS = 64KB (V) + 8KB (P) = 72KB (WGP has 320KB).
// ---------------------------------------------------------------------------
__global__ void __launch_bounds__(256)
mha_flash_kernel(const float* __restrict__ QH, const float* __restrict__ KH,
                 const float* __restrict__ VH, float* __restrict__ O, float scale) {
    extern __shared__ char smem[];
    __bf16* ldsV = (__bf16*)smem;                  // [32 kt][2 f][32 lane][16 j]
    __bf16* ldsP = (__bf16*)(smem + 32 * 2 * 32 * 16 * 2);   // [8 wave][32 lane][16 j]

    int h     = blockIdx.y;
    int tid   = threadIdx.x;
    int lane  = tid & 31;
    int wave  = tid >> 5;
    int qblk  = blockIdx.x * 128;
    int cbase = (qblk / CHUNK) * CHUNK;            // chunk start row

    // ---- stage V tile (1024 keys x 32 dh) into B-fragment layout ----
    for (int d = tid; d < 32 * 2 * 32 * 16; d += 256) {
        int kt  = d >> 10;
        int rem = d & 1023;
        int f   = rem >> 9;
        int l2  = (rem >> 4) & 31;
        int j   = d & 15;
        int key = kt * 32 + koff(l2, j);
        int dh  = f * 16 + (l2 & 15);
        ldsV[d] = (__bf16)VH[(size_t)(cbase + key) * 256 + h * 32 + dh];
    }
    __syncthreads();

    int q0    = qblk + wave * 16;
    int khalf = (lane >> 4) << 3;
    v16bf qf = load_frag_f32(QH + (size_t)(q0 + (lane & 15)) * 256 + h * 32 + khalf, scale);

    v8f acc0 = {}, acc1 = {};
    float rm[8], rl[8];
    for (int r = 0; r < 8; ++r) { rm[r] = -1e30f; rl[r] = 0.0f; }

    __bf16* pw = ldsP + wave * 512;
    int mh = (lane >> 4) << 3;

    for (int kb = 0; kb < CHUNK; kb += 32) {
        // scores: two 16-key tiles, K as column-major B fragment from global
        v16bf kf0 = load_frag_f32(KH + (size_t)(cbase + kb + (lane & 15)) * 256 + h * 32 + khalf, 1.0f);
        v16bf kf1 = load_frag_f32(KH + (size_t)(cbase + kb + 16 + (lane & 15)) * 256 + h * 32 + khalf, 1.0f);
        v8f z = {};
        v8f s0 = WMMA_BF16(qf, kf0, z);
        v8f s1 = WMMA_BF16(qf, kf1, z);

        // online softmax per row (row m = r + 8*(lane>>4), cols across 16 lanes)
        for (int r = 0; r < 8; ++r) {
            float tmx = fmaxf(s0[r], s1[r]);
            tmx = fmaxf(tmx, __shfl_xor(tmx, 1, 32));
            tmx = fmaxf(tmx, __shfl_xor(tmx, 2, 32));
            tmx = fmaxf(tmx, __shfl_xor(tmx, 4, 32));
            tmx = fmaxf(tmx, __shfl_xor(tmx, 8, 32));
            float mnew = fmaxf(rm[r], tmx);
            float p0 = __expf(s0[r] - mnew);
            float p1 = __expf(s1[r] - mnew);
            float ps = p0 + p1;
            ps += __shfl_xor(ps, 1, 32);
            ps += __shfl_xor(ps, 2, 32);
            ps += __shfl_xor(ps, 4, 32);
            ps += __shfl_xor(ps, 8, 32);
            float fac = __expf(rm[r] - mnew);
            rl[r] = rl[r] * fac + ps;
            rm[r] = mnew;
            acc0[r] *= fac;
            acc1[r] *= fac;
            // scatter P (C layout) into A-fragment layout in LDS
            int m  = r + mh;
            int k0 = lane & 15;           // key-in-32 for tile 0
            int k1 = 16 + (lane & 15);    // key-in-32 for tile 1
            int lt0 = m + (((k0 >> 3) & 1) << 4);
            int lt1 = m + (((k1 >> 3) & 1) << 4);
            int j0 = (k0 & 7) + (((k0 >> 4) & 1) << 3);
            int j1 = (k1 & 7) + (((k1 >> 4) & 1) << 3);
            pw[lt0 * 16 + j0] = (__bf16)p0;
            pw[lt1 * 16 + j1] = (__bf16)p1;
        }

        // P (16x32) x V (32x32) accumulated into C operand
        v16bf pf = *(const v16bf*)(pw + lane * 16);
        const v16bf* vb = (const v16bf*)(ldsV) + (size_t)(kb >> 5) * 64 + lane;
        acc0 = WMMA_BF16(pf, vb[0],  acc0);   // dh cols 0..15
        acc1 = WMMA_BF16(pf, vb[32], acc1);   // dh cols 16..31
    }

    for (int r = 0; r < 8; ++r) {
        int m = r + mh;
        float inv = 1.0f / rl[r];
        size_t ro = (size_t)(q0 + m) * 256 + h * 32;
        O[ro + (lane & 15)]      = acc0[r] * inv;
        O[ro + 16 + (lane & 15)] = acc1[r] * inv;
    }
}

// ---------------------------------------------------------------------------
// Host orchestration
// ---------------------------------------------------------------------------
extern "C" void kernel_launch(void* const* d_in, const int* in_sizes, int n_in,
                              void* d_out, int out_size, void* d_ws, size_t ws_size,
                              hipStream_t stream) {
    // ---- inputs (setup_inputs order) ----
    const float* x        = (const float*)d_in[0];
    const int*   Acg_r    = (const int*)d_in[1];
    const int*   Acg_c    = (const int*)d_in[2];
    const float* Acg_v    = (const float*)d_in[3];
    const int*   Ahy_r    = (const int*)d_in[4];
    const int*   Ahy_c    = (const int*)d_in[5];
    const float* Ahy_v    = (const float*)d_in[6];
    const int*   Asg_r    = (const int*)d_in[7];
    const int*   Asg_c    = (const int*)d_in[8];
    const float* Asg_v    = (const float*)d_in[9];
    const int*   Lcg_r    = (const int*)d_in[10];
    const int*   Lcg_c    = (const int*)d_in[11];
    const float* Lcg_v    = (const float*)d_in[12];
    const int*   Lhy_r    = (const int*)d_in[13];
    const int*   Lhy_c    = (const int*)d_in[14];
    const float* Lhy_v    = (const float*)d_in[15];
    const int*   Lsg_r    = (const int*)d_in[16];
    const int*   Lsg_c    = (const int*)d_in[17];
    const float* Lsg_v    = (const float*)d_in[18];
    const float* cg_W     = (const float*)d_in[19];
    const float* cg_b     = (const float*)d_in[20];
    const float* hg_W     = (const float*)d_in[21];
    const float* hg_b     = (const float*)d_in[22];
    const float* sg_W     = (const float*)d_in[23];
    const float* sg_b     = (const float*)d_in[24];
    const float* sn_hyp_W = (const float*)d_in[25];
    const float* sn_hyp_b = (const float*)d_in[26];
    const float* sn_cg_W  = (const float*)d_in[27];
    const float* sn_cg_b  = (const float*)d_in[28];
    const float* sn_sg_W  = (const float*)d_in[29];
    const float* sn_sg_b  = (const float*)d_in[30];
    const float* spa_hyp_W  = (const float*)d_in[31];
    const float* spa_hyp_b  = (const float*)d_in[32];
    const float* spa_star_W = (const float*)d_in[33];
    const float* spa_star_b = (const float*)d_in[34];
    const float* spa_clq_W  = (const float*)d_in[35];
    const float* spa_clq_b  = (const float*)d_in[36];
    const float* spa_in_w   = (const float*)d_in[37];
    const float* spa_in_b   = (const float*)d_in[38];
    const float* spa_out_w  = (const float*)d_in[39];
    const float* spa_out_b  = (const float*)d_in[40];
    const float* spe_hyp_W  = (const float*)d_in[41];
    const float* spe_hyp_b  = (const float*)d_in[42];
    const float* spe_star_W = (const float*)d_in[43];
    const float* spe_star_b = (const float*)d_in[44];
    const float* spe_clq_W  = (const float*)d_in[45];
    const float* spe_clq_b  = (const float*)d_in[46];
    const float* spe_in_w   = (const float*)d_in[47];
    const float* spe_in_b   = (const float*)d_in[48];
    const float* spe_out_w  = (const float*)d_in[49];
    const float* spe_out_b  = (const float*)d_in[50];

    const int ne_n = in_sizes[1];   // 491520
    const int ne_m = in_sizes[7];   // 573440

    float* out = (float*)d_out;

    // ---- workspace carve ----
    char* p = (char*)d_ws;
    const size_t PACK_ELEMS = 65536;                 // per 256x256 weight
    __bf16* Bpack = (__bf16*)p;
    p += 23 * PACK_ELEMS * sizeof(__bf16);
    p = (char*)(((uintptr_t)p + 255) & ~(uintptr_t)255);

    const size_t MROW = (size_t)MM * 256, NROW = (size_t)NN * 256;
    float* S0 = (float*)p; p += MROW * sizeof(float);   // M rows scratch
    float* S1 = (float*)p; p += MROW * sizeof(float);   // M rows scratch
    float* S2 = (float*)p; p += NROW * sizeof(float);
    float* S3 = (float*)p; p += NROW * sizeof(float);
    float* S4 = (float*)p; p += NROW * sizeof(float);
    float* AO = (float*)p; p += NROW * sizeof(float);   // attention output
    float* CG = (float*)p; p += NROW * sizeof(float);   // cg_feat
    float* HG = (float*)p; p += NROW * sizeof(float);   // hg_feat
    float* SG = (float*)p; p += MROW * sizeof(float);   // sg_feat
    float* XH = (float*)p; p += NROW * sizeof(float);   // X_hyp
    float* XC = (float*)p; p += NROW * sizeof(float);   // X_cg
    float* XS = (float*)p; p += MROW * sizeof(float);   // X_sg

    // ---- helpers ----
    auto pack = [&](const float* W, int slot, int tr) {
        pack_weight_kernel<<<256, 256, 0, stream>>>(W, Bpack + (size_t)slot * PACK_ELEMS, tr);
    };
    const size_t GEMM_LDS = 8 * 8 * 32 * 16 * sizeof(__bf16);   // 64 KB
    auto gemm = [&](const float* A, int slot, const float* bias, float* C,
                    int rows, int ldc, int col_off) {
        dim3 g(rows / 128, 2);
        gemm_wmma_kernel<<<g, 256, GEMM_LDS, stream>>>(A, Bpack + (size_t)slot * PACK_ELEMS,
                                                       bias, C, ldc, col_off);
    };
    auto spmm = [&](const int* r, const int* c, const float* v, const float* X,
                    float* Y, int ne) {
        int total = ne * 64;
        spmm_kernel<<<(total + 255) / 256, 256, 0, stream>>>(r, c, v, X, Y, ne);
    };
    auto initb = [&](float* Y, const float* b, int rows) {
        int tot = rows * 256;
        init_bias_kernel<<<(tot + 255) / 256, 256, 0, stream>>>(Y, b, tot);
    };
    auto relu = [&](float* Y, int rows) {
        int tot = rows * 256;
        relu_kernel<<<(tot + 255) / 256, 256, 0, stream>>>(Y, tot);
    };
    const float scale = 1.0f / sqrtf((float)DH);
    const size_t FLASH_LDS = (32 * 2 * 32 * 16) * 2 + (8 * 32 * 16) * 2;  // 73728 B
    auto flash = [&](const float* Q, const float* K, const float* V, float* O) {
        dim3 g(NN / 128, HEADS);
        mha_flash_kernel<<<g, 256, FLASH_LDS, stream>>>(Q, K, V, O, scale);
    };

    // ---- pack all 23 weights (bf16, B-fragment layout) ----
    pack(cg_W, 0, 0);            pack(cg_W + 65536, 1, 0);
    pack(hg_W, 2, 0);            pack(hg_W + 65536, 3, 0);
    pack(sg_W, 4, 0);            pack(sg_W + 65536, 5, 0);
    pack(sn_hyp_W, 6, 1);        pack(sn_cg_W, 7, 1);        pack(sn_sg_W, 8, 1);
    pack(spa_hyp_W, 9, 1);       pack(spa_star_W, 10, 1);    pack(spa_clq_W, 11, 1);
    pack(spa_in_w, 12, 1);       pack(spa_in_w + 65536, 13, 1);
    pack(spa_in_w + 131072, 14, 1);
    pack(spa_out_w, 15, 1);
    pack(spe_hyp_W, 16, 1);      pack(spe_star_W, 17, 1);    pack(spe_clq_W, 18, 1);
    pack(spe_in_w, 19, 1);       pack(spe_in_w + 65536, 20, 1);
    pack(spe_in_w + 131072, 21, 1);
    pack(spe_out_w, 22, 1);

    // ---- expansion nets ----
    // cg:  t0=x@W0 -> S0 ; h=b0+spmm -> S1 ; relu ; t1=h@W1 -> S0 ; CG=b1+spmm
    gemm(x, 0, nullptr, S0, NN, 256, 0);
    initb(S1, cg_b, NN);
    spmm(Acg_r, Acg_c, Acg_v, S0, S1, ne_n);
    relu(S1, NN);
    gemm(S1, 1, nullptr, S0, NN, 256, 0);
    initb(CG, cg_b + 256, NN);
    spmm(Acg_r, Acg_c, Acg_v, S0, CG, ne_n);
    // hg
    gemm(x, 2, nullptr, S0, NN, 256, 0);
    initb(S1, hg_b, NN);
    spmm(Ahy_r, Ahy_c, Ahy_v, S0, S1, ne_n);
    relu(S1, NN);
    gemm(S1, 3, nullptr, S0, NN, 256, 0);
    initb(HG, hg_b + 256, NN);
    spmm(Ahy_r, Ahy_c, Ahy_v, S0, HG, ne_n);
    // sg on x_star = [x ; zeros(EH,256)]
    initb(S1, nullptr, MM);
    hipMemcpyAsync(S1, x, (size_t)NN * 256 * sizeof(float),
                   hipMemcpyDeviceToDevice, stream);
    gemm(S1, 4, nullptr, S0, MM, 256, 0);
    initb(S1, sg_b, MM);
    spmm(Asg_r, Asg_c, Asg_v, S0, S1, ne_m);
    relu(S1, MM);
    gemm(S1, 5, nullptr, S0, MM, 256, 0);
    initb(SG, sg_b + 256, MM);
    spmm(Asg_r, Asg_c, Asg_v, S0, SG, ne_m);

    // X_star / S_star straight into d_out tail (sg_feat is contiguous M x 256)
    hipMemcpyAsync(out + (size_t)NN * 512, SG, (size_t)MM * 256 * sizeof(float),
                   hipMemcpyDeviceToDevice, stream);

    // ---- spectral-node transforms: X = spmm(L, feat @ W.T + b) ----
    gemm(HG, 6, sn_hyp_b, S0, NN, 256, 0);
    initb(XH, nullptr, NN);
    spmm(Lhy_r, Lhy_c, Lhy_v, S0, XH, ne_n);
    gemm(CG, 7, sn_cg_b, S0, NN, 256, 0);
    initb(XC, nullptr, NN);
    spmm(Lcg_r, Lcg_c, Lcg_v, S0, XC, ne_n);
    gemm(SG, 8, sn_sg_b, S0, MM, 256, 0);
    initb(XS, nullptr, MM);
    spmm(Lsg_r, Lsg_c, Lsg_v, S0, XS, ne_m);

    // ---- spectral MHA -> d_out columns [256,512) ----
    gemm(XH, 16, spe_hyp_b,  S2, NN, 256, 0);   // q
    gemm(XC, 17, spe_star_b, S3, NN, 256, 0);   // k
    gemm(XS, 18, spe_clq_b,  S4, NN, 256, 0);   // v (first NN rows of X_sg)
    gemm(S2, 19, spe_in_b,        S0, NN, 256, 0);   // qh
    gemm(S3, 20, spe_in_b + 256,  S1, NN, 256, 0);   // kh
    gemm(S4, 21, spe_in_b + 512,  S2, NN, 256, 0);   // vh (reuse S2)
    flash(S0, S1, S2, AO);
    gemm(AO, 22, spe_out_b, out, NN, 512, 256);

    // ---- spatial MHA -> d_out columns [0,256) ----
    gemm(HG, 9,  spa_hyp_b,  S2, NN, 256, 0);   // q
    gemm(CG, 10, spa_star_b, S3, NN, 256, 0);   // k
    gemm(SG, 11, spa_clq_b,  S4, NN, 256, 0);   // v = X_star @ W.T + b
    gemm(S2, 12, spa_in_b,        S0, NN, 256, 0);
    gemm(S3, 13, spa_in_b + 256,  S1, NN, 256, 0);
    gemm(S4, 14, spa_in_b + 512,  S2, NN, 256, 0);
    flash(S0, S1, S2, AO);
    gemm(AO, 15, spa_out_b, out, NN, 512, 0);

    (void)n_in; (void)out_size; (void)ws_size;
}
